// GraphAttentionLayer_10823317586769
// MI455X (gfx1250) — compile-verified, matrix operand order checked by reading, not run on hardware
//
#include <hip/hip_runtime.h>
#include <stdint.h>

// ---------------------------------------------------------------------------
// GAT layer, fused flash-attention style for MI455X (gfx1250, wave32, WMMA).
//   k1: h = x @ W^T  in bf16 WMMA, stored transposed h_t[B,H,HD,N] (bf16)
//   k2: e_l/e_r = per-head dots, stored [B,H,N] f32
//   k3: per (b, i-tile, head) wave: stream j-tiles, online softmax of
//       lrelu(e_l[i]+e_r[j]) masked by adj, accumulate out^T = V^T x P^T
//       with v_wmma_f32_16x16x32_bf16 (P stays in registers in B-layout).
// HBM floor: adj = 64MB (~2.9us @23.3TB/s); h_t (4MB) + e (256KB) L2-resident.
// ---------------------------------------------------------------------------

typedef __bf16 v16bf __attribute__((ext_vector_type(16)));
typedef float  v8f   __attribute__((ext_vector_type(8)));

#define B_   4
#define N_   2048
#define FIN  128
#define FOUT 256
#define H_   4
#define HD_  64

// native converts -> v_cvt*_bf16 (1 VALU/elt) instead of manual RNE (3 VALU/elt)
__device__ __forceinline__ __bf16 f2bf(float f) { return (__bf16)f; }
__device__ __forceinline__ float bf2f(uint16_t b) {
  return __builtin_bit_cast(float, ((uint32_t)b) << 16);
}

union ABF {                 // one 16x32 (or 32x16) bf16 WMMA operand per lane
  __bf16   b[16];
  uint4    q[2];
  v16bf    v;
};

// ---------------------------------------------------------------------------
// k1: h_t[(b*256+o)*N + n] = bf16( sum_k x[b,n,k] * W[o,k] )
// one wave per 16x16 output tile; A = x rows, B = W rows (A-layout of B^T).
// ---------------------------------------------------------------------------
__global__ void __launch_bounds__(128)
k_gemm_h(const float* __restrict__ x, const float* __restrict__ W,
         uint16_t* __restrict__ h_t) {
  const int wid  = (blockIdx.x * blockDim.x + threadIdx.x) >> 5;
  const int lane = threadIdx.x & 31;
  const int l    = lane & 15;
  const int half = lane >> 4;
  const int kb   = half * 8;

  const int OT    = FOUT / 16;         // 16 o-tiles
  const int itile = wid / OT;          // 0..511
  const int otile = wid % OT;
  const int i0    = itile * 16;
  const int o0    = otile * 16;

  const float* xrow = x + (size_t)(i0 + l) * FIN;
  const float* wrow = W + (size_t)(o0 + l) * FIN;

  v8f acc = {0.f,0.f,0.f,0.f,0.f,0.f,0.f,0.f};
  #pragma unroll
  for (int kk = 0; kk < FIN; kk += 32) {
    float4 xf0 = *(const float4*)(xrow + kk + kb);
    float4 xf1 = *(const float4*)(xrow + kk + kb + 4);
    float4 xf2 = *(const float4*)(xrow + kk + kb + 16);
    float4 xf3 = *(const float4*)(xrow + kk + kb + 20);
    float4 wf0 = *(const float4*)(wrow + kk + kb);
    float4 wf1 = *(const float4*)(wrow + kk + kb + 4);
    float4 wf2 = *(const float4*)(wrow + kk + kb + 16);
    float4 wf3 = *(const float4*)(wrow + kk + kb + 20);
    ABF a, bm;
    const float xa[16] = {xf0.x,xf0.y,xf0.z,xf0.w, xf1.x,xf1.y,xf1.z,xf1.w,
                          xf2.x,xf2.y,xf2.z,xf2.w, xf3.x,xf3.y,xf3.z,xf3.w};
    const float wa[16] = {wf0.x,wf0.y,wf0.z,wf0.w, wf1.x,wf1.y,wf1.z,wf1.w,
                          wf2.x,wf2.y,wf2.z,wf2.w, wf3.x,wf3.y,wf3.z,wf3.w};
    #pragma unroll
    for (int c = 0; c < 16; ++c) { a.b[c] = f2bf(xa[c]); bm.b[c] = f2bf(wa[c]); }
    acc = __builtin_amdgcn_wmma_f32_16x16x32_bf16(false, a.v, false, bm.v,
                                                  (short)0, acc, false, false);
  }
  // C layout: lane holds o = o0 + (lane&15); rows i = i0 + half*8 + v (v=0..7,
  // consecutive) -> one 16B store along N of h_t[B,H*HD,N].
  const int bb = i0 / N_;
  const int n0 = i0 - bb * N_ + half * 8;
  const int o  = o0 + l;
  union { __bf16 b[8]; uint4 q; } st;
  #pragma unroll
  for (int v = 0; v < 8; ++v) st.b[v] = f2bf(acc[v]);
  *(uint4*)(h_t + (size_t)(bb * FOUT + o) * N_ + n0) = st.q;
}

// ---------------------------------------------------------------------------
// k2: el/er[(b*H+h)*N + n] = sum_d h_t[b,h,d,n] * a_{l,r}[h,d]   (coalesced)
// ---------------------------------------------------------------------------
__global__ void __launch_bounds__(256)
k_el_er(const uint16_t* __restrict__ h_t,
        const float* __restrict__ a_l, const float* __restrict__ a_r,
        float* __restrict__ el, float* __restrict__ er) {
  const int t = blockIdx.x * blockDim.x + threadIdx.x;  // (b*H+h)*N + n
  if (t >= B_ * H_ * N_) return;
  const int n  = t & (N_ - 1);
  const int bh = t / N_;
  const int h  = bh & (H_ - 1);
  const uint16_t* hp = h_t + (size_t)bh * HD_ * N_ + n;
  float sl = 0.f, sr = 0.f;
  #pragma unroll 8
  for (int d = 0; d < HD_; ++d) {
    float hv = bf2f(hp[(size_t)d * N_]);
    sl = fmaf(hv, a_l[h * HD_ + d], sl);
    sr = fmaf(hv, a_r[h * HD_ + d], sr);
  }
  el[t] = sl;
  er[t] = sr;
}

// ---------------------------------------------------------------------------
// k3: fused masked softmax + PV.  block = 4 waves = 4 heads of one (b,i-tile).
// ---------------------------------------------------------------------------
__global__ void __launch_bounds__(128)
k_attn(const int* __restrict__ adj, const uint16_t* __restrict__ h_t,
       const float* __restrict__ el, const float* __restrict__ er,
       float* __restrict__ out) {
  const int head = threadIdx.x >> 5;
  const int lane = threadIdx.x & 31;
  const int l    = lane & 15;
  const int half = lane >> 4;
  const int kb   = half * 8;        // this lane's K(j)-halves: {kb..kb+7, kb+16..kb+23}

  const int b     = blockIdx.x / (N_ / 16);
  const int itile = blockIdx.x % (N_ / 16);
  const int i0    = itile * 16;

  const int bh = b * H_ + head;
  const float el_i = el[(size_t)bh * N_ + i0 + l];            // score row i = l
  const float* erp = er + (size_t)bh * N_;
  const int* adjrow = adj + ((size_t)b * N_ + i0 + l) * N_;   // adj[b, i0+l, :]
  const uint16_t* vt = h_t + (size_t)bh * HD_ * N_;           // V^T rows: [d][j]

  const float NEG = -__builtin_inff();
  float m = NEG, lsum = 0.f;
  v8f acc[4];
  #pragma unroll
  for (int ds = 0; ds < 4; ++ds) acc[ds] = (v8f){0.f,0.f,0.f,0.f,0.f,0.f,0.f,0.f};

  #pragma unroll 2
  for (int jt = 0; jt < N_; jt += 32) {
    // prefetch next tile's adj line (the only HBM-bound stream)
    if (jt + 32 < N_) __builtin_prefetch(adjrow + jt + 32 + kb, 0, 1);

    // adjacency + e_r for this lane's 16 (i=l, j) score slots (one 128B line/row)
    int4   a0 = *(const int4*)  (adjrow + jt + kb);
    int4   a1 = *(const int4*)  (adjrow + jt + kb + 4);
    int4   a2 = *(const int4*)  (adjrow + jt + kb + 16);
    int4   a3 = *(const int4*)  (adjrow + jt + kb + 20);
    float4 e0 = *(const float4*)(erp    + jt + kb);
    float4 e1 = *(const float4*)(erp    + jt + kb + 4);
    float4 e2 = *(const float4*)(erp    + jt + kb + 16);
    float4 e3 = *(const float4*)(erp    + jt + kb + 20);
    const int   av[16] = {a0.x,a0.y,a0.z,a0.w, a1.x,a1.y,a1.z,a1.w,
                          a2.x,a2.y,a2.z,a2.w, a3.x,a3.y,a3.z,a3.w};
    const float ev[16] = {e0.x,e0.y,e0.z,e0.w, e1.x,e1.y,e1.z,e1.w,
                          e2.x,e2.y,e2.z,e2.w, e3.x,e3.y,e3.z,e3.w};

    float s[16];
    float mloc = NEG;
    #pragma unroll
    for (int c = 0; c < 16; ++c) {
      float e = el_i + ev[c];
      e = (e > 0.f) ? e : 0.2f * e;           // leaky_relu(0.2)
      s[c] = (av[c] != 0) ? e : NEG;          // mask AFTER lrelu (as reference)
      mloc = fmaxf(mloc, s[c]);
    }
    mloc = fmaxf(mloc, __shfl_xor(mloc, 16, 32));   // combine the two K-halves
    const float mnew  = fmaxf(m, mloc);
    const float scale = (m == mnew) ? 1.f : __expf(m - mnew);

    ABF pb;                                   // P in A-layout-of-P == B-operand of P^T
    float psum = 0.f;
    #pragma unroll
    for (int c = 0; c < 16; ++c) {
      float p = (s[c] == NEG) ? 0.f : __expf(s[c] - mnew);
      psum += p;
      pb.b[c] = f2bf(p);
    }
    psum += __shfl_xor(psum, 16, 32);
    lsum  = lsum * scale + psum;
    m     = mnew;

    // C layout of out^T tile: lane's column is i = lane&15 -> one scalar rescale;
    // skip the 32 multiplies entirely when no row max changed (wave-uniform).
    const float cscale = __shfl(scale, l, 32);
    const bool needScale = __any(cscale != 1.0f);
    #pragma unroll
    for (int ds = 0; ds < 4; ++ds) {
      ABF va;                                 // V^T rows d = ds*16 + l, contiguous j
      const uint16_t* vp = vt + (size_t)(ds * 16 + l) * N_ + jt + kb;
      va.q[0] = *(const uint4*)(vp);
      va.q[1] = *(const uint4*)(vp + 16);
      if (needScale) acc[ds] = acc[ds] * cscale;
      acc[ds] = __builtin_amdgcn_wmma_f32_16x16x32_bf16(false, va.v, false, pb.v,
                                                        (short)0, acc[ds], false, false);
    }
  }

  const float lrow = __shfl(lsum, l, 32);     // row sum for column i = lane&15
  const float inv  = 1.0f / lrow;

  // acc[ds][v] = out[i0+l][head*64 + ds*16 + half*8 + v]
  float* orow = out + ((size_t)b * N_ + i0 + l) * FOUT + head * HD_;
  #pragma unroll
  for (int ds = 0; ds < 4; ++ds) {
    float4 lo = {acc[ds][0]*inv, acc[ds][1]*inv, acc[ds][2]*inv, acc[ds][3]*inv};
    float4 hi = {acc[ds][4]*inv, acc[ds][5]*inv, acc[ds][6]*inv, acc[ds][7]*inv};
    *(float4*)(orow + ds * 16 + half * 8)     = lo;
    *(float4*)(orow + ds * 16 + half * 8 + 4) = hi;
  }
}

// ---------------------------------------------------------------------------
// workspace: h_t bf16 [B*H*HD*N] (4MB) | el f32 [B*H*N] | er f32 [B*H*N]
// ---------------------------------------------------------------------------
extern "C" void kernel_launch(void* const* d_in, const int* in_sizes, int n_in,
                              void* d_out, int out_size, void* d_ws, size_t ws_size,
                              hipStream_t stream) {
  const float* x   = (const float*)d_in[0];
  const int*   adj = (const int*)  d_in[1];
  const float* W   = (const float*)d_in[2];
  const float* a_l = (const float*)d_in[3];
  const float* a_r = (const float*)d_in[4];
  float* out = (float*)d_out;

  uint16_t* h_t = (uint16_t*)d_ws;                       // 2,097,152 bf16 = 4MB
  float*    el  = (float*)((char*)d_ws + (size_t)B_ * FOUT * N_ * 2);
  float*    er  = el + (size_t)B_ * H_ * N_;

  // k1: 8192 tile-waves, 4 waves/block
  k_gemm_h<<<dim3((B_ * N_ / 16) * (FOUT / 16) / 4), dim3(128), 0, stream>>>(x, W, h_t);
  // k2: one thread per (b,h,n)
  k_el_er<<<dim3(B_ * H_ * N_ / 256), dim3(256), 0, stream>>>(h_t, a_l, a_r, el, er);
  // k3: one block per (b, i-tile), 4 head-waves
  k_attn<<<dim3(B_ * (N_ / 16)), dim3(128), 0, stream>>>(adj, h_t, el, er, out);
}
